// SwitchMoELayer_40355512714061
// MI455X (gfx1250) — matrix operand order repaired and checked by reading, not compiled, not run on hardware
//
#include <hip/hip_runtime.h>
#include <hip/hip_bf16.h>

// ---------------- problem constants ----------------
#define NB   4
#define NT   4096
#define ND   1024
#define NF   4096
#define NE   8
#define NTOK (NB * NT)          // 16384
#define CAP  2560               // ceil(16384/8 * 1.25)
#define LNEPS 1e-5f

typedef __attribute__((ext_vector_type(16))) __bf16 bf16x16;
typedef __attribute__((ext_vector_type(8)))  __bf16 bf16x8;
typedef __attribute__((ext_vector_type(4)))  __bf16 bf16x4;
typedef __attribute__((ext_vector_type(8)))  float  f32x8;

union BF16x16U { bf16x16 v; bf16x8 h[2]; };

static __device__ __forceinline__ __bf16 f2bf(float f) {
    return static_cast<__bf16>(f);          // native v_cvt bf16 path
}
static __device__ __forceinline__ float fast_rcp(float f) {
    return __builtin_amdgcn_rcpf(f);        // v_rcp_f32
}
static __device__ __forceinline__ float fast_rsq(float f) {
    return __builtin_amdgcn_rsqf(f);        // v_rsq_f32
}

// ---------------- 1) router: logits, softmax, top-1 ----------------
__global__ void __launch_bounds__(256)
moe_router_kernel(const float* __restrict__ x, const float* __restrict__ rw,
                  int* __restrict__ top_idx, float* __restrict__ top_val,
                  float* __restrict__ sum_prob, float* __restrict__ sum_cnt) {
    int lane  = threadIdx.x & 31;
    int token = blockIdx.x * 8 + (threadIdx.x >> 5);
    if (token >= NTOK) return;
    const float* xr = x + (size_t)token * ND;
    float acc[NE];
#pragma unroll
    for (int e = 0; e < NE; ++e) acc[e] = 0.f;
    for (int d = lane; d < ND; d += 32) {
        float xv = xr[d];
        const float* wr = rw + (size_t)d * NE;
#pragma unroll
        for (int e = 0; e < NE; ++e) acc[e] = fmaf(xv, wr[e], acc[e]);
    }
#pragma unroll
    for (int e = 0; e < NE; ++e)
#pragma unroll
        for (int off = 16; off > 0; off >>= 1)
            acc[e] += __shfl_xor(acc[e], off, 32);
    if (lane == 0) {
        float mx = acc[0];
#pragma unroll
        for (int e = 1; e < NE; ++e) mx = fmaxf(mx, acc[e]);
        float p[NE]; float den = 0.f;
#pragma unroll
        for (int e = 0; e < NE; ++e) { p[e] = __expf(acc[e] - mx); den += p[e]; }
        float inv = fast_rcp(den);
        int best = 0; float bv = p[0] * inv;
#pragma unroll
        for (int e = 0; e < NE; ++e) {
            p[e] *= inv;
            if (p[e] > bv) { bv = p[e]; best = e; }   // strict > keeps first max (argmax semantics)
        }
        top_idx[token] = best;
        top_val[token] = bv;
#pragma unroll
        for (int e = 0; e < NE; ++e) atomicAdd(&sum_prob[e], p[e]);
        atomicAdd(&sum_cnt[best], 1.0f);
    }
}

// ---------------- 2) token-order per-expert prefix positions ----------------
__global__ void __launch_bounds__(256)
moe_scan_kernel(const int* __restrict__ top_idx, int* __restrict__ pos) {
    __shared__ int cnt[256][NE];
    int t = threadIdx.x;
    int base = t * (NTOK / 256);   // 64 tokens per thread
    int c[NE];
#pragma unroll
    for (int e = 0; e < NE; ++e) c[e] = 0;
    for (int i = 0; i < NTOK / 256; ++i) c[top_idx[base + i]]++;
#pragma unroll
    for (int e = 0; e < NE; ++e) cnt[t][e] = c[e];
    __syncthreads();
    if (t < NE) {
        int run = 0;
        for (int j = 0; j < 256; ++j) { int v = cnt[j][t]; cnt[j][t] = run; run += v; }
    }
    __syncthreads();
#pragma unroll
    for (int e = 0; e < NE; ++e) c[e] = cnt[t][e];
    for (int i = 0; i < NTOK / 256; ++i) {
        int e = top_idx[base + i];
        pos[base + i] = c[e]++;
    }
}

// ---------------- 3) dispatch: scatter fp32 -> bf16 expert buffers ----------------
__global__ void __launch_bounds__(256)
moe_dispatch_kernel(const float* __restrict__ x, const int* __restrict__ top_idx,
                    const int* __restrict__ pos, __bf16* __restrict__ buf) {
    int lane  = threadIdx.x & 31;
    int token = blockIdx.x * 8 + (threadIdx.x >> 5);
    if (token >= NTOK) return;
    int p = pos[token];
    if (p >= CAP) return;                       // dropped beyond capacity
    int e = top_idx[token];
    const float* src = x + (size_t)token * ND;
    __bf16* dst = buf + ((size_t)(e * CAP + p)) * ND;
    for (int d = lane * 4; d < ND; d += 128) {
        float4 v = *(const float4*)(src + d);
        bf16x4 o;
        o[0] = f2bf(v.x); o[1] = f2bf(v.y); o[2] = f2bf(v.z); o[3] = f2bf(v.w);
        *(bf16x4*)(dst + d) = o;
    }
}

// ---------------- WMMA fragment helpers (bf16 16x16x32, wave32 ISA layout) ----------------
// A 16x32: lanes 0-15 hold M=lane, K 0-7 & 16-23 ; lanes 16-31 hold M=lane-16, K 8-15 & 24-31
static __device__ __forceinline__ bf16x16 load_a_frag(const __bf16* lds_row0, int row_stride,
                                                      int r0, int lane) {
    int r   = r0 + (lane & 15);
    int klo = (lane >> 4) * 8;
    BF16x16U u;
    u.h[0] = *(const bf16x8*)(lds_row0 + (size_t)r * row_stride + klo);
    u.h[1] = *(const bf16x8*)(lds_row0 + (size_t)r * row_stride + klo + 16);
    return u.v;
}
// B 32x16 stored transposed in LDS as [n][k]: lanes 0-15 -> N=lane,K 0-7 & 16-23 ; lanes 16-31 -> N=lane-16,K 8-15 & 24-31
static __device__ __forceinline__ bf16x16 load_b_frag(const __bf16* lds_n0, int row_stride,
                                                      int c0, int lane) {
    int c  = c0 + (lane & 15);
    int kh = (lane >> 4) * 8;
    BF16x16U u;
    u.h[0] = *(const bf16x8*)(lds_n0 + (size_t)c * row_stride + kh);
    u.h[1] = *(const bf16x8*)(lds_n0 + (size_t)c * row_stride + kh + 16);
    return u.v;
}

#define LDS_KS 40   // 32 K + 8 pad (80B row stride, 16B aligned)

// ---------------- 4) fused gate/up GEMM + SiLU : h = silu(buf@gate)*(buf@up) ----------------
__global__ void __launch_bounds__(256)
moe_gateup_kernel(const __bf16* __restrict__ buf, const float* __restrict__ gate_w,
                  const float* __restrict__ up_w, __bf16* __restrict__ h) {
    const int e  = blockIdx.z;
    const int m0 = blockIdx.y * 64;    // rows within expert
    const int c0 = blockIdx.x * 128;   // cols within F
    __shared__ __align__(16) __bf16 As[64][LDS_KS];
    __shared__ __align__(16) __bf16 Bg[128][LDS_KS];
    __shared__ __align__(16) __bf16 Bu[128][LDS_KS];

    const __bf16* Ag = buf + ((size_t)e * CAP + m0) * ND;
    const float*  Gg = gate_w + (size_t)e * ND * NF;
    const float*  Ug = up_w   + (size_t)e * ND * NF;

    const int tid = threadIdx.x, lane = tid & 31, wid = tid >> 5;
    const int wm = (wid & 1) * 32;      // wave tile: 32 rows
    const int wn = (wid >> 1) * 32;     //            32 cols

    f32x8 accG[2][2] = {};
    f32x8 accU[2][2] = {};

    for (int k0 = 0; k0 < ND; k0 += 32) {
        // stage A tile 64x32 (bf16 direct copy)
        {
            int r = tid >> 2, ks = (tid & 3) * 8;
            bf16x8 v = *(const bf16x8*)(Ag + (size_t)r * ND + k0 + ks);
            *(bf16x8*)&As[r][ks] = v;
            if (k0 + 32 < ND)
                __builtin_prefetch(Ag + (size_t)r * ND + k0 + 32 + ks, 0, 1);
        }
        // stage B tiles 32x128 (fp32 -> bf16, transposed to [n][k])
        {
            int n = tid & 127, kh = (tid >> 7) * 16;
            BF16x16U tg, tu;
#pragma unroll
            for (int i = 0; i < 16; ++i) {
                size_t gi = (size_t)(k0 + kh + i) * NF + c0 + n;
                tg.v[i] = f2bf(Gg[gi]);
                tu.v[i] = f2bf(Ug[gi]);
            }
            *(bf16x8*)&Bg[n][kh]     = tg.h[0];
            *(bf16x8*)&Bg[n][kh + 8] = tg.h[1];
            *(bf16x8*)&Bu[n][kh]     = tu.h[0];
            *(bf16x8*)&Bu[n][kh + 8] = tu.h[1];
        }
        __syncthreads();

        bf16x16 aF[2];
        aF[0] = load_a_frag(&As[0][0], LDS_KS, wm,      lane);
        aF[1] = load_a_frag(&As[0][0], LDS_KS, wm + 16, lane);
#pragma unroll
        for (int j = 0; j < 2; ++j) {
            bf16x16 bG = load_b_frag(&Bg[0][0], LDS_KS, wn + j * 16, lane);
            bf16x16 bU = load_b_frag(&Bu[0][0], LDS_KS, wn + j * 16, lane);
#pragma unroll
            for (int i = 0; i < 2; ++i) {
                accG[i][j] = __builtin_amdgcn_wmma_f32_16x16x32_bf16(
                    false, aF[i], false, bG, (short)0, accG[i][j], false, false);
                accU[i][j] = __builtin_amdgcn_wmma_f32_16x16x32_bf16(
                    false, aF[i], false, bU, (short)0, accU[i][j], false, false);
            }
        }
        __syncthreads();
    }

    // epilogue: h = silu(G) * U  (C/D layout: VGPR v -> M=v (lanes<16) / v+8 (lanes>=16), N=lane&15)
    __bf16* Hg = h + ((size_t)e * CAP + m0) * NF + c0;
#pragma unroll
    for (int i = 0; i < 2; ++i)
#pragma unroll
        for (int j = 0; j < 2; ++j) {
            int rbase = wm + i * 16 + ((lane & 16) ? 8 : 0);
            int col   = wn + j * 16 + (lane & 15);
#pragma unroll
            for (int v = 0; v < 8; ++v) {
                float g = accG[i][j][v], u = accU[i][j][v];
                float s = g * fast_rcp(1.f + __expf(-g));   // v_rcp_f32, no div expansion
                Hg[(size_t)(rbase + v) * NF + col] = f2bf(s * u);
            }
        }
}

// ---------------- 5) down GEMM : y = h @ down_w ----------------
__global__ void __launch_bounds__(256)
moe_down_kernel(const __bf16* __restrict__ h, const float* __restrict__ down_w,
                __bf16* __restrict__ y) {
    const int e  = blockIdx.z;
    const int m0 = blockIdx.y * 64;
    const int c0 = blockIdx.x * 128;   // cols within D
    __shared__ __align__(16) __bf16 As[64][LDS_KS];
    __shared__ __align__(16) __bf16 Bd[128][LDS_KS];

    const __bf16* Ag = h + ((size_t)e * CAP + m0) * NF;
    const float*  Dg = down_w + (size_t)e * NF * ND;

    const int tid = threadIdx.x, lane = tid & 31, wid = tid >> 5;
    const int wm = (wid & 1) * 32;
    const int wn = (wid >> 1) * 32;

    f32x8 acc[2][2] = {};

    for (int k0 = 0; k0 < NF; k0 += 32) {
        {
            int r = tid >> 2, ks = (tid & 3) * 8;
            bf16x8 v = *(const bf16x8*)(Ag + (size_t)r * NF + k0 + ks);
            *(bf16x8*)&As[r][ks] = v;
            if (k0 + 32 < NF)
                __builtin_prefetch(Ag + (size_t)r * NF + k0 + 32 + ks, 0, 1);
        }
        {
            int n = tid & 127, kh = (tid >> 7) * 16;
            BF16x16U td;
#pragma unroll
            for (int i = 0; i < 16; ++i) {
                td.v[i] = f2bf(Dg[(size_t)(k0 + kh + i) * ND + c0 + n]);
            }
            *(bf16x8*)&Bd[n][kh]     = td.h[0];
            *(bf16x8*)&Bd[n][kh + 8] = td.h[1];
        }
        __syncthreads();

        bf16x16 aF[2];
        aF[0] = load_a_frag(&As[0][0], LDS_KS, wm,      lane);
        aF[1] = load_a_frag(&As[0][0], LDS_KS, wm + 16, lane);
#pragma unroll
        for (int j = 0; j < 2; ++j) {
            bf16x16 bD = load_b_frag(&Bd[0][0], LDS_KS, wn + j * 16, lane);
#pragma unroll
            for (int i = 0; i < 2; ++i)
                acc[i][j] = __builtin_amdgcn_wmma_f32_16x16x32_bf16(
                    false, aF[i], false, bD, (short)0, acc[i][j], false, false);
        }
        __syncthreads();
    }

    __bf16* Yg = y + ((size_t)e * CAP + m0) * ND + c0;
#pragma unroll
    for (int i = 0; i < 2; ++i)
#pragma unroll
        for (int j = 0; j < 2; ++j) {
            int rbase = wm + i * 16 + ((lane & 16) ? 8 : 0);
            int col   = wn + j * 16 + (lane & 15);
#pragma unroll
            for (int v = 0; v < 8; ++v)
                Yg[(size_t)(rbase + v) * ND + col] = f2bf(acc[i][j][v]);
        }
}

// ---------------- 6) combine: gather + scale + residual + LayerNorm ----------------
__global__ void __launch_bounds__(256)
moe_combine_ln_kernel(const float* __restrict__ x, const __bf16* __restrict__ y,
                      const int* __restrict__ top_idx, const int* __restrict__ pos,
                      const float* __restrict__ top_val,
                      const float* __restrict__ ln_g, const float* __restrict__ ln_b,
                      float* __restrict__ out) {
    int token = blockIdx.x;
    int tid = threadIdx.x, lane = tid & 31, wid = tid >> 5;
    int e = top_idx[token];
    int p = pos[token];
    bool keep = (p < CAP);
    int pc = keep ? p : 0;
    float tv = keep ? top_val[token] : 0.f;
    const float*  xr = x + (size_t)token * ND;
    const __bf16* yr = y + ((size_t)e * CAP + pc) * ND;

    float vals[4], s = 0.f, s2 = 0.f;
#pragma unroll
    for (int i = 0; i < 4; ++i) {
        int d = tid + i * 256;
        float t = (float)yr[d] * tv;
        float r = t + xr[d];
        vals[i] = r; s += r; s2 += r * r;
    }
#pragma unroll
    for (int off = 16; off > 0; off >>= 1) {
        s  += __shfl_xor(s, off, 32);
        s2 += __shfl_xor(s2, off, 32);
    }
    __shared__ float rs[8], rs2[8];
    if (lane == 0) { rs[wid] = s; rs2[wid] = s2; }
    __syncthreads();
    float ts = 0.f, ts2 = 0.f;
#pragma unroll
    for (int i = 0; i < 8; ++i) { ts += rs[i]; ts2 += rs2[i]; }
    float mean = ts * (1.f / ND);
    float var  = ts2 * (1.f / ND) - mean * mean;
    float rstd = fast_rsq(var + LNEPS);     // v_rsq_f32
    float* orow = out + (size_t)token * ND;
#pragma unroll
    for (int i = 0; i < 4; ++i) {
        int d = tid + i * 256;
        orow[d] = (vals[i] - mean) * rstd * ln_g[d] + ln_b[d];
    }
}

// ---------------- 7) load-balance loss ----------------
__global__ void moe_lb_kernel(const float* __restrict__ sum_prob,
                              const float* __restrict__ sum_cnt,
                              float* __restrict__ loss) {
    if (threadIdx.x == 0 && blockIdx.x == 0) {
        float acc = 0.f;
#pragma unroll
        for (int e = 0; e < NE; ++e)
            acc += (sum_cnt[e] * (1.f / NTOK)) * (sum_prob[e] * (1.f / NTOK));
        loss[0] = (float)NE * acc;
    }
}

// ---------------- launch ----------------
extern "C" void kernel_launch(void* const* d_in, const int* in_sizes, int n_in,
                              void* d_out, int out_size, void* d_ws, size_t ws_size,
                              hipStream_t stream) {
    const float* x        = (const float*)d_in[0];
    const float* router_w = (const float*)d_in[1];
    const float* gate_w   = (const float*)d_in[2];
    const float* up_w     = (const float*)d_in[3];
    const float* down_w   = (const float*)d_in[4];
    const float* ln_g     = (const float*)d_in[5];
    const float* ln_b     = (const float*)d_in[6];

    char* w = (char*)d_ws;
    const size_t bufB = (size_t)NE * CAP * ND * sizeof(__bf16);  // 41.9 MB
    const size_t hB   = (size_t)NE * CAP * NF * sizeof(__bf16);  // 167.8 MB
    const size_t yB   = (size_t)NE * CAP * ND * sizeof(__bf16);  // 41.9 MB
    __bf16* buf  = (__bf16*)w;
    __bf16* h    = (__bf16*)(w + bufB);
    __bf16* y    = (__bf16*)(w + bufB + hB);
    int*    tidx = (int*)   (w + bufB + hB + yB);
    int*    tpos = (int*)   (w + bufB + hB + yB + NTOK * 4);
    float*  tval = (float*) (w + bufB + hB + yB + NTOK * 8);
    float*  sprb = (float*) (w + bufB + hB + yB + NTOK * 12);
    float*  scnt = sprb + NE;

    float* out  = (float*)d_out;
    float* loss = out + (size_t)NTOK * ND;

    hipMemsetAsync(buf, 0, bufB, stream);            // dropped / unfilled slots must be zero
    hipMemsetAsync(sprb, 0, 2 * NE * sizeof(float), stream);

    moe_router_kernel<<<NTOK / 8, 256, 0, stream>>>(x, router_w, tidx, tval, sprb, scnt);
    moe_scan_kernel<<<1, 256, 0, stream>>>(tidx, tpos);
    moe_dispatch_kernel<<<NTOK / 8, 256, 0, stream>>>(x, tidx, tpos, buf);
    moe_gateup_kernel<<<dim3(NF / 128, CAP / 64, NE), 256, 0, stream>>>(buf, gate_w, up_w, h);
    moe_down_kernel<<<dim3(ND / 128, CAP / 64, NE), 256, 0, stream>>>(h, down_w, y);
    moe_combine_ln_kernel<<<NTOK, 256, 0, stream>>>(x, y, tidx, tpos, tval, ln_g, ln_b, out);
    moe_lb_kernel<<<1, 32, 0, stream>>>(sprb, scnt, loss);
}
